// Decoder_22737556865467
// MI455X (gfx1250) — compile-verified, matrix operand order checked by reading, not run on hardware
//
#include <hip/hip_runtime.h>
#include <math.h>

// ---------------- CDNA5 WMMA types ----------------
typedef __bf16 bf16_t;
typedef __attribute__((ext_vector_type(16))) __bf16 v16bf;
typedef __attribute__((ext_vector_type(8)))  __bf16 v8bf;
typedef __attribute__((ext_vector_type(8)))  float  v8f;

#define VOCAB 32000
#define EMB   256
#define HID   2048
#define TT    80
#define BB    32

// LDS tile: 128 rows x 32 K, padded to 40 bf16 (80 B) per row to spread banks.
#define BK      32
#define LDS_LDA 40

// ---- bf16 16x16x32 fragment loader (global, row-major [rows, ld]) ----
// 16-bit A/B layout (ISA 7.12.2): lanes 0-15 row r, K = {0..7, 16..23};
// lanes 16-31 row r, K = {8..15, 24..31}. Two 16B loads per lane.
__device__ __forceinline__ v16bf load_frag(const bf16_t* __restrict__ p, int ld, int lane) {
  const int sub = lane >> 4;
  const bf16_t* base = p + (size_t)(lane & 15) * (size_t)ld + sub * 8;
  v8bf lo = *(const v8bf*)(base);
  v8bf hi = *(const v8bf*)(base + 16);
  v16bf f;
#pragma unroll
  for (int i = 0; i < 8; ++i) { f[i] = lo[i]; f[8 + i] = hi[i]; }
  return f;
}

// ---- same fragment load but from an LDS tile with row stride LDS_LDA ----
__device__ __forceinline__ v16bf lds_frag(const bf16_t* base, int lane) {
  const int sub = lane >> 4;
  const bf16_t* p = base + (lane & 15) * LDS_LDA + sub * 8;
  v8bf lo = *(const v8bf*)(p);        // ds_load_b128
  v8bf hi = *(const v8bf*)(p + 16);   // ds_load_b128
  v16bf f;
#pragma unroll
  for (int i = 0; i < 8; ++i) { f[i] = lo[i]; f[8 + i] = hi[i]; }
  return f;
}

__device__ __forceinline__ v8f wmma_bf16(v16bf a, v16bf b, v8f c) {
  return __builtin_amdgcn_wmma_f32_16x16x32_bf16(false, a, false, b, (short)0, c,
                                                 false, false);
}

__device__ __forceinline__ float sigf(float x) { return 1.f / (1.f + __expf(-x)); }

// Generic LDS addresses carry the LDS byte offset in the low 32 bits
// (aperture in [63:32]); truncation yields the DS address for async ops.
__device__ __forceinline__ unsigned lds_addr_of(const void* p) {
  return (unsigned)(size_t)p;
}

// Issue one 16B async global->LDS copy (ASYNCcnt-tracked DMA, no VGPR data).
__device__ __forceinline__ void async_copy_b128(void* lds_dst, const void* gsrc) {
  asm volatile("global_load_async_to_lds_b128 %0, %1, off"
               :: "v"(lds_addr_of(lds_dst)), "v"((unsigned long long)(size_t)gsrc)
               : "memory");
}

__device__ __forceinline__ void wait_async0() {
  asm volatile("s_wait_asynccnt 0x0" ::: "memory");
}

// Copy a 128x32 bf16 tile (global row-major, leading dim ldK) into LDS.
// 512 x 16B chunks spread over 256 threads (2 per thread).
__device__ __forceinline__ void copy_tile_async(bf16_t* dst, const bf16_t* __restrict__ src,
                                                int ldK, int tid) {
#pragma unroll
  for (int i = 0; i < 2; ++i) {
    const int c = tid + i * 256;
    const int row = c >> 2, ch = c & 3;
    async_copy_b128(dst + row * LDS_LDA + ch * 8,
                    src + (size_t)row * ldK + ch * 8);
  }
}

// ---------------- fp32 [K,N] -> bf16 [N,K] transposed convert ----------------
__global__ void __launch_bounds__(256)
k_transpose_bf16(const float* __restrict__ in, bf16_t* __restrict__ out, int K, int N) {
  __shared__ float tile[32][33];
  const int n0 = blockIdx.x * 32, k0 = blockIdx.y * 32;
  const int tx = threadIdx.x, ty = threadIdx.y;  // (32, 8)
#pragma unroll
  for (int i = 0; i < 32; i += 8)
    tile[ty + i][tx] = in[(size_t)(k0 + ty + i) * N + n0 + tx];
  __syncthreads();
#pragma unroll
  for (int i = 0; i < 32; i += 8)
    out[(size_t)(n0 + ty + i) * K + k0 + tx] = (bf16_t)tile[tx][ty + i];
}

// ---------------- teacher-forced embedding gather -> bf16 X ----------------
__global__ void __launch_bounds__(EMB)
k_embed(const int* __restrict__ trg, const float* __restrict__ emb,
        bf16_t* __restrict__ Xb) {
  const int r = blockIdx.x;             // 0 .. T*B-1
  const int t = r / BB, b = r % BB;
  const int word = (t == 0) ? 0 : trg[(t - 1) * BB + b];   // <unk> = 0
  Xb[(size_t)r * EMB + threadIdx.x] = (bf16_t)emb[(size_t)word * EMB + threadIdx.x];
}

// ---------------- LDS-staged bf16 GEMM + bias ----------------
// C[M,N] = A[M,K] @ Bt[N,K]^T + bias.  Block tile 128x128, BK=32,
// double-buffered async global->LDS staging; 8 waves in 2(M) x 4(N),
// wave tile 64x32 = 4x2 fragments -> 8 WMMAs per K-step per wave.
// All 12 ds_load_b128 are issued before the WMMA burst so the scheduler can
// use partial s_wait_dscnt thresholds instead of full drains.
__global__ void __launch_bounds__(256)
k_gemm_bias(const bf16_t* __restrict__ A, const bf16_t* __restrict__ Bt,
            const float* __restrict__ bias, float* __restrict__ C,
            int M, int N, int K) {
  __shared__ __align__(16) bf16_t sA[2][128 * LDS_LDA];
  __shared__ __align__(16) bf16_t sB[2][128 * LDS_LDA];

  const int tid  = threadIdx.x;
  const int lane = tid & 31;
  const int wave = tid >> 5;
  const int wm = (wave >> 2) * 64;   // 0 or 64
  const int wn = (wave & 3) * 32;    // 0,32,64,96
  const int bm = blockIdx.y * 128;
  const int bn = blockIdx.x * 128;

  const bf16_t* Ag = A + (size_t)bm * K;
  const bf16_t* Bg = Bt + (size_t)bn * K;

  // prime buffer 0
  copy_tile_async(sA[0], Ag, K, tid);
  copy_tile_async(sB[0], Bg, K, tid);
  wait_async0();
  __syncthreads();

  v8f acc[4][2] = {};
  const int nk = K / BK;
  for (int kt = 0; kt < nk; ++kt) {
    const int cur = kt & 1;
    if (kt + 1 < nk) {  // overlap next tile's DMA with this tile's WMMAs
      copy_tile_async(sA[cur ^ 1], Ag + (size_t)(kt + 1) * BK, K, tid);
      copy_tile_async(sB[cur ^ 1], Bg + (size_t)(kt + 1) * BK, K, tid);
    }
    // ---- load phase: all fragments first (in-order DS returns allow the
    // scheduler to overlap the first WMMAs with the tail DS loads) ----
    v16bf b0 = lds_frag(&sB[cur][(wn +  0) * LDS_LDA], lane);
    v16bf b1 = lds_frag(&sB[cur][(wn + 16) * LDS_LDA], lane);
    v16bf a[4];
#pragma unroll
    for (int i = 0; i < 4; ++i)
      a[i] = lds_frag(&sA[cur][(wm + i * 16) * LDS_LDA], lane);
    // ---- math phase: 8 back-to-back WMMAs ----
#pragma unroll
    for (int i = 0; i < 4; ++i) {
      acc[i][0] = wmma_bf16(a[i], b0, acc[i][0]);
      acc[i][1] = wmma_bf16(a[i], b1, acc[i][1]);
    }
    wait_async0();       // own async copies of buffer cur^1 landed
    __syncthreads();     // everyone done reading cur / writing cur^1
  }

  // C/D layout: VGPR j -> row (j + 8*(lane>=16)), col = lane&15.
  const int hi8 = (lane >> 4) * 8;
  const int cin = lane & 15;
#pragma unroll
  for (int i = 0; i < 4; ++i)
#pragma unroll
    for (int j = 0; j < 2; ++j) {
      const int col = bn + wn + j * 16 + cin;
      const float bv = bias[col];
      const int row0 = bm + wm + i * 16 + hi8;
#pragma unroll
      for (int r = 0; r < 8; ++r)
        C[(size_t)(row0 + r) * N + col] = acc[i][j][r] + bv;
    }
}

// ---------------- copy initial hidden state (fp32 + bf16) ----------------
__global__ void __launch_bounds__(256)
k_init_h(const float* __restrict__ hidden, float* __restrict__ h0,
         bf16_t* __restrict__ hb0) {
  const int i = blockIdx.x * 256 + threadIdx.x;   // BB*HID = 65536
  const float v = hidden[i];
  h0[i] = v;
  hb0[i] = (bf16_t)v;
}

// ---------------- one recurrence step, fully fused ----------------
// G_w = h@W_w, G_lw = h@W_lw for a 64-column slab + the gated elementwise
// update for those columns; writes h_{t+1} (fp32 ping-pong) and bf16 directly
// into the H matrix consumed by the hoisted output GEMM. Direct loads: the
// 16.8 MB fused weight set is L2-resident across the 80 sequential launches.
__global__ void __launch_bounds__(128)
k_step(const bf16_t* __restrict__ Ab,    // [32, HID] bf16 h_t
       const float*  __restrict__ h_in,  // [32, HID] fp32 h_t
       const bf16_t* __restrict__ Wwt,   // [HID, HID] (transposed, bf16)
       const bf16_t* __restrict__ Wlwt,  // [HID, HID]
       const float*  __restrict__ b_w,
       const float*  __restrict__ b_lw,
       const float*  __restrict__ FEATt, // [32, HID] = x_t@W_lin + b_lin
       const float*  __restrict__ LFEt,  // [32, HID] = x_t@W_ll  + b_ll
       float*  __restrict__ h_out,       // [32, HID] fp32 h_{t+1}
       bf16_t* __restrict__ Hb_t) {      // [32, HID] bf16 h_{t+1} (into H)
  const int lane = threadIdx.x & 31;
  const int wave = threadIdx.x >> 5;
  const int n0 = blockIdx.x * 64 + wave * 16;

  v8f cw[2] = {}, cl[2] = {};
  const bf16_t* Bw = Wwt + (size_t)n0 * HID;
  const bf16_t* Bl = Wlwt + (size_t)n0 * HID;
#pragma unroll 2
  for (int k = 0; k < HID; k += 32) {
    v16bf a0 = load_frag(Ab + k, HID, lane);
    v16bf a1 = load_frag(Ab + (size_t)16 * HID + k, HID, lane);
    v16bf bw = load_frag(Bw + k, HID, lane);
    v16bf bl = load_frag(Bl + k, HID, lane);
    cw[0] = wmma_bf16(a0, bw, cw[0]);
    cw[1] = wmma_bf16(a1, bw, cw[1]);
    cl[0] = wmma_bf16(a0, bl, cl[0]);
    cl[1] = wmma_bf16(a1, bl, cl[1]);
  }
  const int col = n0 + (lane & 15);
  const float bwv = b_w[col];
  const float blv = b_lw[col];
  const int hi8 = (lane >> 4) * 8;
#pragma unroll
  for (int i = 0; i < 2; ++i) {
#pragma unroll
    for (int r = 0; r < 8; ++r) {
      const int m = i * 16 + hi8 + r;
      const size_t idx = (size_t)m * HID + col;
      const float feat = FEATt[idx];
      const float lfe  = LFEt[idx];
      const float hprv = h_in[idx];
      const float nval = tanhf(cw[i][r] + bwv + feat);
      const float l    = sigf(feat) * sigf(cl[i][r] + blv + lfe);
      const float hn   = hprv + l * (nval - hprv);
      h_out[idx] = hn;
      Hb_t[idx]  = (bf16_t)hn;
    }
  }
}

// ---------------- host-side launcher ----------------
extern "C" void kernel_launch(void* const* d_in, const int* in_sizes, int n_in,
                              void* d_out, int out_size, void* d_ws, size_t ws_size,
                              hipStream_t stream) {
  const float* hidden = (const float*)d_in[0];
  const int*   trg    = (const int*)d_in[1];
  const float* emb    = (const float*)d_in[2];
  const float* W_lin  = (const float*)d_in[3];
  const float* b_lin  = (const float*)d_in[4];
  const float* W_w    = (const float*)d_in[5];
  const float* b_w    = (const float*)d_in[6];
  const float* W_lw   = (const float*)d_in[7];
  const float* b_lw   = (const float*)d_in[8];
  const float* W_ll   = (const float*)d_in[9];
  const float* b_ll   = (const float*)d_in[10];
  const float* W_out  = (const float*)d_in[11];
  const float* b_out  = (const float*)d_in[12];
  float* out = (float*)d_out;   // [T, B, VOCAB] fp32

  // -------- workspace carve (256B aligned), ~204 MB total --------
  char* ws = (char*)d_ws;
  auto carve = [&](size_t bytes) -> char* {
    char* p = ws;
    ws += (bytes + 255) & ~(size_t)255;
    return p;
  };
  bf16_t* Wlin_t = (bf16_t*)carve((size_t)HID * EMB * 2);     // [HID, EMB]
  bf16_t* Wll_t  = (bf16_t*)carve((size_t)HID * EMB * 2);
  bf16_t* Wwt    = (bf16_t*)carve((size_t)HID * HID * 2);     // [HID, HID]
  bf16_t* Wlwt   = (bf16_t*)carve((size_t)HID * HID * 2);
  bf16_t* Wout_t = (bf16_t*)carve((size_t)VOCAB * HID * 2);   // [VOCAB, HID], 131 MB -> L2
  bf16_t* Xb     = (bf16_t*)carve((size_t)TT * BB * EMB * 2); // [T*B, EMB]
  float*  FEAT   = (float*) carve((size_t)TT * BB * HID * 4); // [T*B, HID]
  float*  LFE    = (float*) carve((size_t)TT * BB * HID * 4);
  bf16_t* Hb     = (bf16_t*)carve((size_t)TT * BB * HID * 2); // all h_{t+1}, bf16
  bf16_t* hb0    = (bf16_t*)carve((size_t)BB * HID * 2);
  float*  hping  = (float*) carve((size_t)BB * HID * 4);
  float*  hpong  = (float*) carve((size_t)BB * HID * 4);
  (void)ws_size; (void)in_sizes; (void)n_in; (void)out_size;

  const dim3 tb(32, 8);
  // fp32 -> bf16 transposed weight copies (once per launch; W_out streams 262MB)
  k_transpose_bf16<<<dim3(HID / 32, EMB / 32), tb, 0, stream>>>(W_lin, Wlin_t, EMB, HID);
  k_transpose_bf16<<<dim3(HID / 32, EMB / 32), tb, 0, stream>>>(W_ll,  Wll_t,  EMB, HID);
  k_transpose_bf16<<<dim3(HID / 32, HID / 32), tb, 0, stream>>>(W_w,   Wwt,    HID, HID);
  k_transpose_bf16<<<dim3(HID / 32, HID / 32), tb, 0, stream>>>(W_lw,  Wlwt,   HID, HID);
  k_transpose_bf16<<<dim3(VOCAB / 32, HID / 32), tb, 0, stream>>>(W_out, Wout_t, HID, VOCAB);

  // teacher-forced embedding gather for all timesteps
  k_embed<<<TT * BB, EMB, 0, stream>>>(trg, emb, Xb);

  // batched input-side projections: FEAT = X@W_lin + b_lin, LFE = X@W_ll + b_ll
  k_gemm_bias<<<dim3(HID / 128, (TT * BB) / 128), 256, 0, stream>>>(
      Xb, Wlin_t, b_lin, FEAT, TT * BB, HID, EMB);
  k_gemm_bias<<<dim3(HID / 128, (TT * BB) / 128), 256, 0, stream>>>(
      Xb, Wll_t, b_ll, LFE, TT * BB, HID, EMB);

  // initial state
  k_init_h<<<(BB * HID) / 256, 256, 0, stream>>>(hidden, hping, hb0);

  // sequential recurrence: one fused kernel per step (weights L2-resident)
  float* hbufs[2] = {hping, hpong};
  for (int t = 0; t < TT; ++t) {
    const bf16_t* Ab = (t == 0) ? hb0 : (Hb + (size_t)(t - 1) * BB * HID);
    k_step<<<HID / 64, 128, 0, stream>>>(
        Ab, hbufs[t & 1], Wwt, Wlwt, b_w, b_lw,
        FEAT + (size_t)t * BB * HID, LFE + (size_t)t * BB * HID,
        hbufs[(t + 1) & 1], Hb + (size_t)t * BB * HID);
  }

  // hoisted output projection: [2560, 2048] @ [2048, 32000] + b_out, 335 GFLOP
  k_gemm_bias<<<dim3(VOCAB / 128, (TT * BB) / 128), 256, 0, stream>>>(
      Hb, Wout_t, b_out, out, TT * BB, VOCAB, HID);
}
